// Reins_28810640621884
// MI455X (gfx1250) — matrix operand order repaired and checked by reading, not compiled, not run on hardware
//
#include <hip/hip_runtime.h>
#include <hip/hip_bf16.h>

// ---------- types ----------
typedef __attribute__((ext_vector_type(16))) __bf16 v16bf;
typedef __attribute__((ext_vector_type(8)))  __bf16 v8bf;
typedef __attribute__((ext_vector_type(8)))  float  v8f;

#define NTOK 1024   // tokens in x (feats[1:])
#define BB   32
#define CC   1024
#define HH   128
#define MT   32     // tokens per block (2 M-subtiles of 16)
#define MSUB (MT / 16)

// ---------- helpers ----------
__device__ __forceinline__ unsigned short f2bfu(float f) {
    unsigned u = __builtin_bit_cast(unsigned, f);
    unsigned r = u + 0x7FFFu + ((u >> 16) & 1u);
    return (unsigned short)(r >> 16);
}
__device__ __forceinline__ float bfu2f(unsigned short s) {
    return __builtin_bit_cast(float, ((unsigned)s) << 16);
}
__device__ __forceinline__ float gelu_f(float x) {
    return 0.5f * x * (1.0f + erff(x * 0.70710678118654752f));
}
__device__ __forceinline__ float hash_uniform(unsigned idx, unsigned salt) {
    unsigned x = idx * 2654435761u + salt;
    x ^= x >> 16; x *= 0x7feb352du;
    x ^= x >> 15; x *= 0x846ca68bu;
    x ^= x >> 16;
    return (float)(x >> 8) * (1.0f / 16777216.0f);
}
// assemble v16bf from two 16B chunks (A-frag halves or contiguous B-frag)
__device__ __forceinline__ v16bf ld2x8(const unsigned short* p0, const unsigned short* p1) {
    v8bf lo = *reinterpret_cast<const v8bf*>(p0);
    v8bf hi = *reinterpret_cast<const v8bf*>(p1);
    v16bf r;
#pragma unroll
    for (int i = 0; i < 8; i++) { r[i] = lo[i]; r[i + 8] = hi[i]; }
    return r;
}
__device__ __forceinline__ v16bf ld_b16(const unsigned short* p) { return ld2x8(p, p + 8); }

// ---------- k0: weight convert / transpose to bf16 ----------
__global__ __launch_bounds__(256) void k0_convert(const float* __restrict__ Wd,
                                                  const float* __restrict__ Wu,
                                                  const float* __restrict__ Wt,
                                                  const float* __restrict__ Wf,
                                                  unsigned short* __restrict__ WdT,
                                                  unsigned short* __restrict__ WuT,
                                                  unsigned short* __restrict__ WtT,
                                                  unsigned short* __restrict__ WfT) {
    int idx = blockIdx.x * 256 + threadIdx.x;
    if (idx >= HH * CC) return;
    int h = idx >> 10, c = idx & (CC - 1);       // (H,C) layout
    WdT[h * CC + c] = f2bfu(Wd[idx]);
    WtT[h * CC + c] = f2bfu(Wt[idx]);
    int c2 = idx >> 7, h2 = idx & (HH - 1);      // (C,H) layout
    WuT[c2 * HH + h2] = f2bfu(Wu[idx]);
    WfT[c2 * HH + h2] = f2bfu(Wf[idx]);
}

// ---------- cls row copy ----------
__global__ __launch_bounds__(256) void k_cls(const float* __restrict__ feats, float* __restrict__ out) {
    int idx = blockIdx.x * 256 + threadIdx.x;
    if (idx < BB * CC) out[idx] = feats[idx];
}

// ---------- k1: t_feats = gelu(x Wd^T + bd) Wu^T + bu  (bf16 WMMA, 2x B-reuse) ----------
__global__ __launch_bounds__(256) void k1_tfeats(const float* __restrict__ feats,
                                                 const unsigned short* __restrict__ WdT,
                                                 const float* __restrict__ b_featd,
                                                 const unsigned short* __restrict__ WuT,
                                                 const float* __restrict__ b_featu,
                                                 unsigned short* __restrict__ t_ws) {
    __shared__ unsigned short xs[MT * CC];   // 64 KB
    __shared__ unsigned short hs[MT * HH];   // 8 KB
    const int tile = blockIdx.x, b = blockIdx.y;
    const int tid = threadIdx.x;

    // stage x tile (MT tokens x 1024 ch) as bf16; one fully-coalesced row per iter
    {
        const int col = tid * 4;
#pragma unroll 4
        for (int r = 0; r < MT; r++) {
            const float4 v = *reinterpret_cast<const float4*>(
                feats + (size_t)(1 + tile * MT + r) * (BB * CC) + (size_t)b * CC + col);
            unsigned short* dst = xs + r * CC + col;
            dst[0] = f2bfu(v.x); dst[1] = f2bfu(v.y); dst[2] = f2bfu(v.z); dst[3] = f2bfu(v.w);
        }
    }
    __syncthreads();

    const int wave = tid >> 5, lane = tid & 31;
    const int mrow = lane & 15;
    const int khalf = (lane & 16) ? 8 : 0;
    const int drow0 = (lane & 16) ? 8 : 0;  // C/D: lanes 16-31 hold M=8..15
    const int n = lane & 15;

    // GEMM1: h = gelu(x Wd^T + bd), each wave owns one 16-wide H tile; B reused over MSUB
    {
        const int h0 = wave * 16;
        v8f acc[MSUB];
#pragma unroll
        for (int ms = 0; ms < MSUB; ms++) acc[ms] = (v8f){};
        for (int k0 = 0; k0 < CC; k0 += 32) {
            v16bf bfrag = ld2x8(WdT + (size_t)(h0 + n) * CC + k0 + khalf, WdT + (size_t)(h0 + n) * CC + k0 + khalf + 16);
#pragma unroll
            for (int ms = 0; ms < MSUB; ms++) {
                const unsigned short* ap = xs + (ms * 16 + mrow) * CC + k0 + khalf;
                v16bf a = ld2x8(ap, ap + 16);
                acc[ms] = __builtin_amdgcn_wmma_f32_16x16x32_bf16(false, a, false, bfrag,
                                                                  (short)0, acc[ms], false, false);
            }
        }
        float bias = b_featd[h0 + n];
#pragma unroll
        for (int ms = 0; ms < MSUB; ms++)
#pragma unroll
            for (int i = 0; i < 8; i++)
                hs[(ms * 16 + i + drow0) * HH + h0 + n] = f2bfu(gelu_f(acc[ms][i] + bias));
    }
    __syncthreads();

    // GEMM2: t = h Wu^T + bu, each wave owns 8 C tiles; B reused over MSUB
    unsigned short* tdst = t_ws + ((size_t)b << 20);
    for (int j = 0; j < 8; j++) {
        const int c0 = (wave * 8 + j) * 16;
        v8f acc[MSUB];
#pragma unroll
        for (int ms = 0; ms < MSUB; ms++) acc[ms] = (v8f){};
#pragma unroll
        for (int k0 = 0; k0 < HH; k0 += 32) {
            v16bf bfrag = ld2x8(WuT + (size_t)(c0 + n) * HH + k0 + khalf, WuT + (size_t)(c0 + n) * HH + k0 + khalf + 16);
#pragma unroll
            for (int ms = 0; ms < MSUB; ms++) {
                const unsigned short* ap = hs + (ms * 16 + mrow) * HH + k0 + khalf;
                v16bf a = ld2x8(ap, ap + 16);
                acc[ms] = __builtin_amdgcn_wmma_f32_16x16x32_bf16(false, a, false, bfrag,
                                                                  (short)0, acc[ms], false, false);
            }
        }
        float bias = b_featu[c0 + n];
#pragma unroll
        for (int ms = 0; ms < MSUB; ms++)
#pragma unroll
            for (int i = 0; i < 8; i++) {
                int tok = tile * MT + ms * 16 + i + drow0;
                tdst[(size_t)tok * CC + c0 + n] = f2bfu(acc[ms][i] + bias);
            }
    }
}

// ---------- k2: per-batch selection -> patch mask ----------
__device__ __forceinline__ float bred_max(float v, float* red, int t) {
    red[t] = v; __syncthreads();
    for (int s = 512; s > 0; s >>= 1) { if (t < s) red[t] = fmaxf(red[t], red[t + s]); __syncthreads(); }
    float r = red[0]; __syncthreads(); return r;
}
__device__ __forceinline__ float bred_sum(float v, float* red, int t) {
    red[t] = v; __syncthreads();
    for (int s = 512; s > 0; s >>= 1) { if (t < s) red[t] += red[t + s]; __syncthreads(); }
    float r = red[0]; __syncthreads(); return r;
}
__device__ __forceinline__ int bred_cnt(int pred, int* cnt, int t) {
    __syncthreads();
    if (t == 0) *cnt = 0;
    __syncthreads();
    unsigned long long bal = __ballot(pred);
    if ((t & 31) == 0) atomicAdd(cnt, (int)__popcll(bal));
    __syncthreads();
    return *cnt;
}
__global__ __launch_bounds__(1024) void k2_select(const unsigned short* __restrict__ t_ws,
                                                  float* __restrict__ pmask) {
    __shared__ float red[1024];
    __shared__ int cnt;
    const int b = blockIdx.x, t = threadIdx.x;

    const unsigned short* row = t_ws + ((size_t)b << 20) + (size_t)t * CC;
    float s = 0.0f;
    for (int c = 0; c < CC; c += 8) {
        uint4 v = *reinterpret_cast<const uint4*>(row + c);
        s += bfu2f((unsigned short)(v.x & 0xffff)) + bfu2f((unsigned short)(v.x >> 16));
        s += bfu2f((unsigned short)(v.y & 0xffff)) + bfu2f((unsigned short)(v.y >> 16));
        s += bfu2f((unsigned short)(v.z & 0xffff)) + bfu2f((unsigned short)(v.z >> 16));
        s += bfu2f((unsigned short)(v.w & 0xffff)) + bfu2f((unsigned short)(v.w >> 16));
    }
    float a = s * (1.0f / (float)CC);

    float mx = bred_max(a, red, t);
    float w = __expf(a - mx);
    float sum = bred_sum(w, red, t);
    w /= sum;

    float lo = 0.0f, hi = 1.0f;
    for (int it = 0; it < 28; it++) {
        float mid = 0.5f * (lo + hi);
        int c = bred_cnt(w <= mid ? 1 : 0, &cnt, t);
        if (c < 410) lo = mid; else hi = mid;
    }
    float promoted = (w > hi) ? 1.0f : 0.0f;

    float noise = hash_uniform((unsigned)(b * NTOK + t), 0x9e3779b9u) + promoted;
    lo = 0.0f; hi = 2.0f;
    for (int it = 0; it < 28; it++) {
        float mid = 0.5f * (lo + hi);
        int c = bred_cnt(noise <= mid ? 1 : 0, &cnt, t);
        if (c < 409) lo = mid; else hi = mid;
    }
    pmask[b * NTOK + t] = (noise <= hi) ? 0.0f : 1.0f;
}

// ---------- k3: delta path + residual ----------
__global__ __launch_bounds__(256) void k3_delta(const float* __restrict__ feats,
                                                const float* __restrict__ logits_p,
                                                const float* __restrict__ scale_p,
                                                const unsigned short* __restrict__ t_ws,
                                                const float* __restrict__ pmask,
                                                const unsigned short* __restrict__ WtT,
                                                const float* __restrict__ b_t2f,
                                                const unsigned short* __restrict__ WfT,
                                                const float* __restrict__ b_df,
                                                float* __restrict__ out) {
    __shared__ unsigned short xs[MT * CC];   // in2 tile bf16, 64 KB
    __shared__ unsigned short hs[MT * HH];   // 8 KB
    const int tile = blockIdx.x, b = blockIdx.y;
    const int tid = threadIdx.x;
    const float scale = scale_p[0];
    const float l = logits_p[0] * 2.0f;      // logits / 0.5
    const float eps = 1.1920929e-07f;

    // assemble in2 = x*(1+m) + t_feats*patch_mask  (m from hash-uniform logistic)
    {
        const int col = tid * 4;
        for (int r = 0; r < MT; r++) {
            const int tok = tile * MT + r;
            const float pm = pmask[b * NTOK + tok];
            const float4 v = *reinterpret_cast<const float4*>(
                feats + (size_t)(1 + tok) * (BB * CC) + (size_t)b * CC + col);
            const unsigned short* tsrc = t_ws + ((size_t)b << 20) + (size_t)tok * CC + col;
            unsigned short* dst = xs + r * CC + col;
            const unsigned base = (unsigned)((tok * BB + b) * CC + col);
            float xv[4] = { v.x, v.y, v.z, v.w };
#pragma unroll
            for (int e = 0; e < 4; e++) {
                float u = hash_uniform(base + e, 0xc2b2ae35u);
                u = fminf(fmaxf(u, eps), 1.0f - eps);
                float m = 1.0f / (1.0f + __expf(-(l + __logf(u) - log1pf(-u))));
                float tv = bfu2f(tsrc[e]);
                dst[e] = f2bfu(xv[e] * (1.0f + m) + tv * pm);
            }
        }
    }
    __syncthreads();

    const int wave = tid >> 5, lane = tid & 31;
    const int mrow = lane & 15;
    const int khalf = (lane & 16) ? 8 : 0;
    const int drow0 = (lane & 16) ? 8 : 0;
    const int n = lane & 15;

    // GEMM3: h2 = gelu(in2 Wt2f^T + b); B reused over MSUB
    {
        const int h0 = wave * 16;
        v8f acc[MSUB];
#pragma unroll
        for (int ms = 0; ms < MSUB; ms++) acc[ms] = (v8f){};
        for (int k0 = 0; k0 < CC; k0 += 32) {
            v16bf bfrag = ld2x8(WtT + (size_t)(h0 + n) * CC + k0 + khalf, WtT + (size_t)(h0 + n) * CC + k0 + khalf + 16);
#pragma unroll
            for (int ms = 0; ms < MSUB; ms++) {
                const unsigned short* ap = xs + (ms * 16 + mrow) * CC + k0 + khalf;
                v16bf a = ld2x8(ap, ap + 16);
                acc[ms] = __builtin_amdgcn_wmma_f32_16x16x32_bf16(false, a, false, bfrag,
                                                                  (short)0, acc[ms], false, false);
            }
        }
        float bias = b_t2f[h0 + n];
#pragma unroll
        for (int ms = 0; ms < MSUB; ms++)
#pragma unroll
            for (int i = 0; i < 8; i++)
                hs[(ms * 16 + i + drow0) * HH + h0 + n] = f2bfu(gelu_f(acc[ms][i] + bias));
    }
    __syncthreads();

    // GEMM4: delta = h2 Wdf^T + b ; out = x + delta*scale ; B reused over MSUB
    for (int j = 0; j < 8; j++) {
        const int c0 = (wave * 8 + j) * 16;
        v8f acc[MSUB];
#pragma unroll
        for (int ms = 0; ms < MSUB; ms++) acc[ms] = (v8f){};
#pragma unroll
        for (int k0 = 0; k0 < HH; k0 += 32) {
            v16bf bfrag = ld2x8(WfT + (size_t)(c0 + n) * HH + k0 + khalf, WfT + (size_t)(c0 + n) * HH + k0 + khalf + 16);
#pragma unroll
            for (int ms = 0; ms < MSUB; ms++) {
                const unsigned short* ap = hs + (ms * 16 + mrow) * HH + k0 + khalf;
                v16bf a = ld2x8(ap, ap + 16);
                acc[ms] = __builtin_amdgcn_wmma_f32_16x16x32_bf16(false, a, false, bfrag,
                                                                  (short)0, acc[ms], false, false);
            }
        }
        float bias = b_df[c0 + n];
#pragma unroll
        for (int ms = 0; ms < MSUB; ms++)
#pragma unroll
            for (int i = 0; i < 8; i++) {
                int tok = tile * MT + ms * 16 + i + drow0;
                size_t oidx = (size_t)(1 + tok) * (BB * CC) + (size_t)b * CC + c0 + n;
                out[oidx] = feats[oidx] + (acc[ms][i] + bias) * scale;
            }
    }
}

// ---------- launch ----------
extern "C" void kernel_launch(void* const* d_in, const int* in_sizes, int n_in,
                              void* d_out, int out_size, void* d_ws, size_t ws_size,
                              hipStream_t stream) {
    const float* feats   = (const float*)d_in[0];
    const float* logits  = (const float*)d_in[1];
    const float* scale   = (const float*)d_in[2];
    const float* W_featd = (const float*)d_in[3];
    const float* b_featd = (const float*)d_in[4];
    const float* W_featu = (const float*)d_in[5];
    const float* b_featu = (const float*)d_in[6];
    const float* W_t2f   = (const float*)d_in[7];
    const float* b_t2f   = (const float*)d_in[8];
    const float* W_df    = (const float*)d_in[9];
    const float* b_df    = (const float*)d_in[10];
    float* out = (float*)d_out;

    char* ws = (char*)d_ws;
    size_t o = 0;
    unsigned short* WdT = (unsigned short*)(ws + o); o += (size_t)HH * CC * 2;
    unsigned short* WuT = (unsigned short*)(ws + o); o += (size_t)HH * CC * 2;
    unsigned short* WtT = (unsigned short*)(ws + o); o += (size_t)HH * CC * 2;
    unsigned short* WfT = (unsigned short*)(ws + o); o += (size_t)HH * CC * 2;
    float* pmask        = (float*)(ws + o);          o += (size_t)BB * NTOK * 4;
    o = (o + 255) & ~(size_t)255;
    unsigned short* t_ws = (unsigned short*)(ws + o);  // 64 MB, bf16 [B][NTOK][C]

    k0_convert<<<(HH * CC + 255) / 256, 256, 0, stream>>>(W_featd, W_featu, W_t2f, W_df,
                                                          WdT, WuT, WtT, WfT);
    k_cls<<<(BB * CC + 255) / 256, 256, 0, stream>>>(feats, out);
    k1_tfeats<<<dim3(NTOK / MT, BB), 256, 0, stream>>>(feats, WdT, b_featd, WuT, b_featu, t_ws);
    k2_select<<<BB, 1024, 0, stream>>>(t_ws, pmask);
    k3_delta<<<dim3(NTOK / MT, BB), 256, 0, stream>>>(feats, logits, scale, t_ws, pmask,
                                                      WtT, b_t2f, WfT, b_df, out);
}
